// AutoformerEncoder_48034914239190
// MI455X (gfx1250) — matrix operand order repaired and checked by reading, not compiled, not run on hardware
//
#include <hip/hip_runtime.h>
#include <math.h>

// ---------------- problem constants (match reference) ----------------
#define BDIM   16
#define LSEQ   4096
#define LOG2L  12
#define CDIM   512
#define TOPK   16
#define KSIZE  25
#define KPAD   12
#define NTOT   ((size_t)BDIM * LSEQ * CDIM)
#define PI_F   3.14159265358979323846f

typedef __attribute__((ext_vector_type(16))) __bf16 v16bf;
typedef __attribute__((ext_vector_type(8)))  __bf16 v8bf;
typedef __attribute__((ext_vector_type(8)))  float  v8f;
typedef int v4i __attribute__((vector_size(4 * sizeof(int))));

#define ASG __attribute__((address_space(1)))
#define ASL __attribute__((address_space(3)))

#if defined(__has_builtin)
#if __has_builtin(__builtin_amdgcn_global_load_async_to_lds_b128) && \
    __has_builtin(__builtin_amdgcn_s_wait_asynccnt)
#define HAVE_ASYNC 1
#endif
#endif
#ifndef HAVE_ASYNC
#define HAVE_ASYNC 0
#endif

// =====================================================================
// 1) Series decomposition: trend = 25-tap moving avg (reflect pad),
//    seasonal = x - trend.  One block per (b,l); contiguous over C.
//    x (128MiB) is L2-resident (192MB) so window re-reads are L2 hits.
// =====================================================================
__global__ __launch_bounds__(256)
void decompose_kernel(const float* __restrict__ x,
                      float* __restrict__ trend,
                      float* __restrict__ seasonal) {
  const int bl = blockIdx.x;           // b*LSEQ + l
  const int l  = bl & (LSEQ - 1);
  const int b  = bl >> LOG2L;
  const float* xb = x + (size_t)b * LSEQ * CDIM;
  for (int c = threadIdx.x; c < CDIM; c += 256) {
    float s = 0.f;
#pragma unroll
    for (int k = 0; k < KSIZE; ++k) {
      int idx = l + k - KPAD;                       // reflect (no edge dup)
      idx = idx < 0 ? -idx : idx;
      idx = idx >= LSEQ ? 2 * LSEQ - 2 - idx : idx;
      s += xb[(size_t)idx * CDIM + c];
    }
    const float t = s * (1.0f / KSIZE);
    const size_t off = (size_t)bl * CDIM + c;
    trend[off]    = t;
    seasonal[off] = x[off] - t;
  }
}

// =====================================================================
// 2) FFT autocorrelation + top-k lag mask.  One 256-thread block per
//    (b,c) series; 4096-pt complex FFT resident in LDS (32KB of 320KB).
//    corr = IFFT(|FFT(x)|^2)/N; keep top-16 |corr| lags; emit bf16
//    (it feeds the bf16 WMMA GEMM directly -> half the A-side traffic).
// =====================================================================
__device__ __forceinline__ void fft_stages(float* re, float* im, int tid, float sign) {
#pragma unroll 1
  for (int s = 0; s < LOG2L; ++s) {
    const int   half  = 1 << s;
    const float astep = sign * PI_F / (float)half;  // = sign*2*pi/len
    for (int j = tid; j < (LSEQ >> 1); j += 256) {
      const int k  = j & (half - 1);
      const int i0 = ((j >> s) << (s + 1)) + k;
      const int i1 = i0 + half;
      float wi, wr;
      __sincosf(astep * (float)k, &wi, &wr);
      const float xr = re[i1], xi = im[i1];
      const float tr = wr * xr - wi * xi;
      const float ti = wr * xi + wi * xr;
      const float ur = re[i0], ui = im[i0];
      re[i0] = ur + tr; im[i0] = ui + ti;
      re[i1] = ur - tr; im[i1] = ui - ti;
    }
    __syncthreads();
  }
}

__global__ __launch_bounds__(256)
void autocorr_kernel(const float* __restrict__ seasonal,
                     __bf16* __restrict__ out) {
  __shared__ float re[LSEQ];
  __shared__ float im[LSEQ];
  __shared__ float redv[256];
  __shared__ int   redi[256];

  const int tid = threadIdx.x;
  const int b   = blockIdx.x / CDIM;
  const int c   = blockIdx.x % CDIM;
  const float* col = seasonal + (size_t)b * LSEQ * CDIM + c;

  for (int i = tid; i < LSEQ; i += 256) {           // bit-reversed load
    const int r = (int)(__brev((unsigned)i) >> (32 - LOG2L));
    re[r] = col[(size_t)i * CDIM];
    im[r] = 0.f;
  }
  __syncthreads();
  fft_stages(re, im, tid, -1.f);                    // forward FFT

  for (int i = tid; i < LSEQ; i += 256) {           // power spectrum
    const float a = re[i], bb = im[i];
    re[i] = a * a + bb * bb;
  }
  __syncthreads();
  for (int i = tid; i < LSEQ; i += 256) {           // in-place bit-rev, im := 0
    const int r = (int)(__brev((unsigned)i) >> (32 - LOG2L));
    if (i < r) { const float t = re[i]; re[i] = re[r]; re[r] = t; }
    im[i] = 0.f;
  }
  __syncthreads();
  fft_stages(re, im, tid, +1.f);                    // inverse FFT (unscaled)

  for (int i = tid; i < LSEQ; i += 256) {           // corr + working |corr|
    const float v = re[i] * (1.0f / LSEQ);
    re[i] = v;
    im[i] = fabsf(v);
  }
  __syncthreads();

#pragma unroll 1
  for (int r = 0; r < TOPK; ++r) {                  // 16 rounds of argmax
    float bv = -1.f; int bi = 0;
    for (int i = tid; i < LSEQ; i += 256) {
      const float v = im[i];
      if (v > bv) { bv = v; bi = i; }
    }
    redv[tid] = bv; redi[tid] = bi;
    __syncthreads();
    for (int s = 128; s > 0; s >>= 1) {
      if (tid < s && redv[tid + s] > redv[tid]) {
        redv[tid] = redv[tid + s]; redi[tid] = redi[tid + s];
      }
      __syncthreads();
    }
    if (tid == 0) im[redi[0]] = -1.f;               // select + exclude
    __syncthreads();
  }

  __bf16* ocol = out + (size_t)b * LSEQ * CDIM + c;
  for (int i = tid; i < LSEQ; i += 256)
    ocol[(size_t)i * CDIM] = (im[i] < 0.f) ? (__bf16)re[i] : (__bf16)0.f;
}

// =====================================================================
// 2b) W -> bf16, transposed to N-major [n][k] so the GEMM B tile is a
//     straight row-copy (same geometry as the A tile).  1MB, one-shot.
// =====================================================================
__global__ __launch_bounds__(256)
void wconv_kernel(const float* __restrict__ W, __bf16* __restrict__ Wt) {
  const int n = blockIdx.x;
  for (int k = threadIdx.x; k < CDIM; k += 256)
    Wt[(size_t)n * CDIM + k] = (__bf16)W[(size_t)k * CDIM + n];
}

// =====================================================================
// 3) GEMM + bias: Y[M,512] = A[M,512](bf16) @ Wt[n][k](bf16)^T + b.
//    256 threads = 8 waves; 128x128 tile, BK=32; waves 4(M)x2(N),
//    each wave owns 2x4 16x16 f32 accumulators.
//    Both LDS tiles are row-major-in-K with 40-half (80B) row stride:
//    every fragment = two 16B ds_load_b128, bank-conflict-free.
//    Global->LDS staging via GLOBAL_LOAD_ASYNC_TO_LDS_B128 (ASYNCcnt).
// =====================================================================
#define BM  128
#define BN  128
#define BK  32
#define LDT 40    // halves/row: 80B, 16B-aligned rows, stride 20 DWORDs

__global__ __launch_bounds__(256)
void gemm_bias_kernel(const __bf16* __restrict__ A,   // [M, 512] bf16
                      const __bf16* __restrict__ Wt,  // [512(n), 512(k)] bf16
                      const float* __restrict__ bias,
                      float* __restrict__ Y) {
  __shared__ __bf16 __attribute__((aligned(16))) As[BM * LDT];
  __shared__ __bf16 __attribute__((aligned(16))) Bs[BN * LDT];

  const int tid  = threadIdx.x;
  const int lane = tid & 31;
  const int wave = tid >> 5;
  const int m0 = blockIdx.x * BM;
  const int n0 = blockIdx.y * BN;
  const int wm = (wave & 3) * 32;     // wave tile: 32 rows (M)
  const int wn = (wave >> 2) * 64;    //            64 cols (N)

  const v8f vzero = {0.f, 0.f, 0.f, 0.f, 0.f, 0.f, 0.f, 0.f};
  v8f acc[2][4];
#pragma unroll
  for (int i = 0; i < 2; ++i)
#pragma unroll
    for (int j = 0; j < 4; ++j) acc[i][j] = vzero;

  // stager: 128 rows x 4 16B-segments per tile; thread t -> rows r, r+64
  const int srow = tid >> 2;            // 0..63
  const int sseg = (tid & 3) * 8;       // half-offset: 0,8,16,24

  union U { v16bf v; v8bf h[2]; };

#pragma unroll 1
  for (int k0 = 0; k0 < CDIM; k0 += BK) {
    const __bf16* ag0 = A  + (size_t)(m0 + srow) * CDIM + k0 + sseg;
    const __bf16* ag1 = ag0 + (size_t)64 * CDIM;
    const __bf16* bg0 = Wt + (size_t)(n0 + srow) * CDIM + k0 + sseg;
    const __bf16* bg1 = bg0 + (size_t)64 * CDIM;
    __bf16* al0 = &As[srow * LDT + sseg];
    __bf16* al1 = &As[(srow + 64) * LDT + sseg];
    __bf16* bl0 = &Bs[srow * LDT + sseg];
    __bf16* bl1 = &Bs[(srow + 64) * LDT + sseg];
#if HAVE_ASYNC
    __builtin_amdgcn_global_load_async_to_lds_b128((ASG v4i*)(void*)ag0,
                                                   (ASL v4i*)(void*)al0, 0, 0);
    __builtin_amdgcn_global_load_async_to_lds_b128((ASG v4i*)(void*)ag1,
                                                   (ASL v4i*)(void*)al1, 0, 0);
    __builtin_amdgcn_global_load_async_to_lds_b128((ASG v4i*)(void*)bg0,
                                                   (ASL v4i*)(void*)bl0, 0, 0);
    __builtin_amdgcn_global_load_async_to_lds_b128((ASG v4i*)(void*)bg1,
                                                   (ASL v4i*)(void*)bl1, 0, 0);
    __builtin_amdgcn_s_wait_asynccnt(0);
#else
    *(v8bf*)al0 = *(const v8bf*)ag0;
    *(v8bf*)al1 = *(const v8bf*)ag1;
    *(v8bf*)bl0 = *(const v8bf*)bg0;
    *(v8bf*)bl1 = *(const v8bf*)bg1;
#endif
    if (k0 + BK < CDIM) {                 // global_prefetch_b8 next tiles
      __builtin_prefetch(ag0 + BK, 0, 3);
      __builtin_prefetch(bg0 + BK, 0, 3);
    }
    __syncthreads();

    // A fragments (ISA 7.12.2): lane-half 0 -> K kh+0..7 | kh+16..23, kh=0;
    // lane-half 1 -> kh=8.  Two contiguous 16B runs -> 2x ds_load_b128.
    const int kh = (lane >> 4) * 8;
    v16bf af[2];
#pragma unroll
    for (int mt = 0; mt < 2; ++mt) {
      const __bf16* ap = &As[(wm + mt * 16 + (lane & 15)) * LDT];
      U u;
      u.h[0] = *(const v8bf*)(ap + kh);
      u.h[1] = *(const v8bf*)(ap + kh + 16);
      af[mt] = u.v;
    }
    // B fragments: lane(&15) = N column; same K order as A: lane-half 0 ->
    // K 0-7 | 16-23, lane-half 1 -> K 8-15 | 24-31.  2x ds_load_b128.
    const int kb = (lane >> 4) * 8;
#pragma unroll
    for (int nt = 0; nt < 4; ++nt) {
      const __bf16* bp = &Bs[(wn + nt * 16 + (lane & 15)) * LDT + kb];
      U u;
      u.h[0] = *(const v8bf*)(bp);
      u.h[1] = *(const v8bf*)(bp + 16);
      const v16bf bfv = u.v;
#pragma unroll
      for (int mt = 0; mt < 2; ++mt)
        acc[mt][nt] = __builtin_amdgcn_wmma_f32_16x16x32_bf16(
            false, af[mt], false, bfv, (short)0, acc[mt][nt], false, false);
    }
    __syncthreads();
  }

  // C/D layout: VGPR i -> M = (lane>>4)*8 + i, N = lane&15
  const int nlo = lane & 15;
  const int mhi = (lane >> 4) * 8;
#pragma unroll
  for (int mt = 0; mt < 2; ++mt)
#pragma unroll
    for (int nt = 0; nt < 4; ++nt) {
      const int n = n0 + wn + nt * 16 + nlo;
      const float bv = bias[n];
      float* yp = Y + (size_t)(m0 + wm + mt * 16 + mhi) * CDIM + n;
#pragma unroll
      for (int i = 0; i < 8; ++i)
        yp[(size_t)i * CDIM] = acc[mt][nt][i] + bv;
    }
}

// =====================================================================
// 4) y = LayerNorm(a + r) * scale + bias.  One block per row of 512.
// =====================================================================
__global__ __launch_bounds__(256)
void resln_kernel(const float* __restrict__ a, const float* __restrict__ r,
                  const float* __restrict__ scale, const float* __restrict__ bias,
                  float* __restrict__ y) {
  __shared__ float red[256];
  const int tid = threadIdx.x;
  const size_t row = (size_t)blockIdx.x * CDIM;
  const float v0 = a[row + tid]       + r[row + tid];
  const float v1 = a[row + tid + 256] + r[row + tid + 256];
  red[tid] = v0 + v1;
  __syncthreads();
  for (int s = 128; s > 0; s >>= 1) {
    if (tid < s) red[tid] += red[tid + s];
    __syncthreads();
  }
  const float mean = red[0] * (1.0f / CDIM);
  __syncthreads();
  const float d0 = v0 - mean, d1 = v1 - mean;
  red[tid] = d0 * d0 + d1 * d1;
  __syncthreads();
  for (int s = 128; s > 0; s >>= 1) {
    if (tid < s) red[tid] += red[tid + s];
    __syncthreads();
  }
  const float rstd = rsqrtf(red[0] * (1.0f / CDIM) + 1e-6f);
  y[row + tid]       = d0 * rstd * scale[tid]       + bias[tid];
  y[row + tid + 256] = d1 * rstd * scale[tid + 256] + bias[tid + 256];
}

// =====================================================================
// launch: decompose -> 2x(autocorr, wconv, gemm, resln) -> final resln
// d_out doubles as the `seasonal` buffer (fully overwritten each call).
// workspace: trend(f32) | ff(f32) | auto(bf16) | Wt(bf16)  ~ 336 MiB
// =====================================================================
extern "C" void kernel_launch(void* const* d_in, const int* in_sizes, int n_in,
                              void* d_out, int out_size, void* d_ws, size_t ws_size,
                              hipStream_t stream) {
  const float* x       = (const float*)d_in[0];
  const float* ffw[2]  = {(const float*)d_in[1], (const float*)d_in[5]};
  const float* ffbv[2] = {(const float*)d_in[2], (const float*)d_in[6]};
  const float* lns[2]  = {(const float*)d_in[3], (const float*)d_in[7]};
  const float* lnb[2]  = {(const float*)d_in[4], (const float*)d_in[8]};
  const float* flns    = (const float*)d_in[9];
  const float* flnb    = (const float*)d_in[10];

  float*  seasonal = (float*)d_out;
  float*  trend    = (float*)d_ws;
  float*  ffo      = trend + NTOT;
  __bf16* autob    = (__bf16*)(ffo + NTOT);
  __bf16* wt       = autob + NTOT;

  decompose_kernel<<<BDIM * LSEQ, 256, 0, stream>>>(x, trend, seasonal);

  for (int layer = 0; layer < 2; ++layer) {
    autocorr_kernel<<<BDIM * CDIM, 256, 0, stream>>>(seasonal, autob);
    wconv_kernel<<<CDIM, 256, 0, stream>>>(ffw[layer], wt);
    gemm_bias_kernel<<<dim3((BDIM * LSEQ) / BM, CDIM / BN), 256, 0, stream>>>(
        autob, wt, ffbv[layer], ffo);
    resln_kernel<<<BDIM * LSEQ, 256, 0, stream>>>(seasonal, ffo,
                                                  lns[layer], lnb[layer], seasonal);
  }
  resln_kernel<<<BDIM * LSEQ, 256, 0, stream>>>(seasonal, trend, flns, flnb, seasonal);
}
